// MultiHeadAttention_89532888252726
// MI455X (gfx1250) — matrix operand order repaired and checked by reading, not compiled, not run on hardware
//
#include <hip/hip_runtime.h>
#include <hip/hip_bf16.h>

// ---------------------------------------------------------------------------
// MI455X (gfx1250) fused multi-head attention forward.
// wave32, V_WMMA_F32_16X16X32_F16 for all matrix math, f32 accumulate.
// All GEMM operands pre-converted to f16 so inner loops are pure b128 loads
// (no dependent cvt chains -> loads can stay in flight).
// ---------------------------------------------------------------------------

typedef __attribute__((ext_vector_type(16))) _Float16 v16h;
typedef __attribute__((ext_vector_type(8)))  _Float16 v8h;
typedef __attribute__((ext_vector_type(8)))  float    v8f;
typedef __attribute__((ext_vector_type(4)))  float    v4f;

#define D_MODEL 1024
#define NUM_HEADS 16
#define HEAD_DIM 64
#define BATCH 2
#define SEQ 2048
#define ROWS (BATCH * SEQ)          // 4096
#define BH (BATCH * NUM_HEADS)      // 32

static __device__ __forceinline__ v8f wmma_f16(v16h a, v16h b, v8f c) {
  return __builtin_amdgcn_wmma_f32_16x16x32_f16(false, a, false, b, (short)0, c,
                                                false, false);
}

// Assemble an A-operand v16h from two contiguous 8-half chunks (chunk at p,
// chunk at p+16 halves) -- matches the 16-bit A-matrix 16x32 layout:
// lane<16 : M=lane,    halves = K[8h..8h+7] then K[16+8h..16+8h+7] with h=0
// lane>=16: M=lane-16, same with h=1   (caller passes p = row + 8*half)
static __device__ __forceinline__ v16h ld_a16(const _Float16* p) {
  v8h lo = *(const v8h*)p;
  v8h hi = *(const v8h*)(p + 16);
  v16h r;
#pragma unroll
  for (int i = 0; i < 8; ++i) { r[i] = lo[i]; r[8 + i] = hi[i]; }
  return r;
}

// ---------------------------------------------------------------------------
// Elementwise f32 -> f16 convert (vectorized, grid-stride). Bandwidth-bound
// pre-pass so GEMM inner loops never touch f32 operands.
// ---------------------------------------------------------------------------
__global__ __launch_bounds__(256) void cvt_f16_kernel(
    const float* __restrict__ in, _Float16* __restrict__ out, int n4) {
  for (int i = blockIdx.x * blockDim.x + threadIdx.x; i < n4;
       i += gridDim.x * blockDim.x) {
    v4f x = *(const v4f*)(in + (size_t)i * 4);
    _Float16* o = out + (size_t)i * 4;
    o[0] = (_Float16)x[0];
    o[1] = (_Float16)x[1];
    o[2] = (_Float16)x[2];
    o[3] = (_Float16)x[3];
  }
}

// ---------------------------------------------------------------------------
// Projection: Y = X @ W^T + bias. X f16 [4096x1024], W f16 [1024x1024].
// Wave tile 32 rows x 64 cols: 2 A-sets x 4 B-tiles = 8 WMMAs per K-step.
// Writes f16 head-split layout:
//   transpose_v == 0 : out[bh][s][dh]   (q, k)
//   transpose_v == 1 : out[bh][dh][s]   (v, transposed for attention B-ops)
// ---------------------------------------------------------------------------
__global__ __launch_bounds__(256) void mha_proj_kernel(
    const _Float16* __restrict__ X, const _Float16* __restrict__ W,
    const float* __restrict__ bias, _Float16* __restrict__ out,
    int transpose_v) {
  const int wave = threadIdx.x >> 5;
  const int lane = threadIdx.x & 31;
  const int half = lane >> 4;
  const int l15  = lane & 15;
  const int tile = blockIdx.x * 8 + wave;          // 2048 tiles
  const int m0 = (tile & 127) * 32;                // 128 row tiles (32 rows)
  const int n0 = (tile >> 7) * 64;                 // 16 col tiles (64 cols)

  const _Float16* arow0 = X + (size_t)(m0 + l15) * D_MODEL + 8 * half;
  const _Float16* arow1 = arow0 + (size_t)16 * D_MODEL;
  const _Float16* wrow  = W + (size_t)(n0 + l15) * D_MODEL + 8 * half;

  v8f c00 = {}, c01 = {}, c02 = {}, c03 = {};
  v8f c10 = {}, c11 = {}, c12 = {}, c13 = {};

  for (int k0 = 0; k0 < D_MODEL; k0 += 32) {
    // Batch-issue all operand loads (12x b128), then compute.
    v16h a0 = ld_a16(arow0 + k0);
    v16h a1 = ld_a16(arow1 + k0);
    v16h b0 = ld_a16(wrow + 0 * 16 * D_MODEL + k0);
    v16h b1 = ld_a16(wrow + 1 * 16 * D_MODEL + k0);
    v16h b2 = ld_a16(wrow + 2 * 16 * D_MODEL + k0);
    v16h b3 = ld_a16(wrow + 3 * 16 * D_MODEL + k0);
    c00 = wmma_f16(a0, b0, c00);
    c01 = wmma_f16(a0, b1, c01);
    c02 = wmma_f16(a0, b2, c02);
    c03 = wmma_f16(a0, b3, c03);
    c10 = wmma_f16(a1, b0, c10);
    c11 = wmma_f16(a1, b1, c11);
    c12 = wmma_f16(a1, b2, c12);
    c13 = wmma_f16(a1, b3, c13);
  }

  // Epilogue: C layout -> VGPR i holds (row = i + 8*half, col = 16t + l15)
#pragma unroll
  for (int r = 0; r < 2; ++r) {
#pragma unroll
    for (int t = 0; t < 4; ++t) {
      v8f c;
      if (r == 0) c = (t == 0) ? c00 : (t == 1) ? c01 : (t == 2) ? c02 : c03;
      else        c = (t == 0) ? c10 : (t == 1) ? c11 : (t == 2) ? c12 : c13;
      const int col = n0 + 16 * t + l15;
      const float bv = bias[col];
      const int h = col >> 6, d = col & 63;
#pragma unroll
      for (int i = 0; i < 8; ++i) {
        const int row = m0 + 16 * r + i + 8 * half;  // global row [0,4096)
        const int b = row >> 11, s = row & 2047;
        const float val = c[i] + bv;
        if (transpose_v)
          out[((size_t)(b * NUM_HEADS + h) * HEAD_DIM + d) * SEQ + s] =
              (_Float16)val;
        else
          out[((size_t)(b * NUM_HEADS + h) * SEQ + s) * HEAD_DIM + d] =
              (_Float16)val;
      }
    }
  }
}

// ---------------------------------------------------------------------------
// Fused attention: per wave one (bh, 16-q-row) block. Computes S^T = K @ Q^T
// tiles so softmax rows are in-lane and P lands directly in A-operand layout.
// Streams raw scores to d_out, online softmax, P@V accumulation in registers.
// ---------------------------------------------------------------------------
__global__ __launch_bounds__(256) void mha_attn_kernel(
    const _Float16* __restrict__ qbuf,   // [bh][s][64]
    const _Float16* __restrict__ kbuf,   // [bh][s][64]
    const _Float16* __restrict__ vT,     // [bh][64][s]
    _Float16* __restrict__ attn,         // [b][s][1024] (heads recombined)
    float* __restrict__ scores) {        // [bh][s][s]
  const int wave = threadIdx.x >> 5;
  const int lane = threadIdx.x & 31;
  const int half = lane >> 4;
  const int l15  = lane & 15;
  const int unit = blockIdx.x * 8 + wave;          // 4096 units
  const int bh = unit >> 7;                        // 0..31
  const int q0 = (unit & 127) * 16;

  const _Float16* qbh = qbuf + (size_t)bh * SEQ * HEAD_DIM;
  const _Float16* kbh = kbuf + (size_t)bh * SEQ * HEAD_DIM;
  const _Float16* vbh = vT   + (size_t)bh * HEAD_DIM * SEQ;
  float* sc = scores + (size_t)bh * SEQ * SEQ + (size_t)q0 * SEQ;

  // Q as B operand (N = q row = l15, K = dh), two K-groups covering dh=64.
  const _Float16* qp = qbh + (size_t)(q0 + l15) * HEAD_DIM + 8 * half;
  const v16h bq0 = ld_a16(qp);        // dh 0..31 group
  const v16h bq1 = ld_a16(qp + 32);   // dh 32..63 group

  float m_run = -1e30f, l_run = 0.f;
  v8f o0 = {}, o1 = {}, o2 = {}, o3 = {};
  const float scale = 0.125f;                 // 1/sqrt(64)

  for (int j0 = 0; j0 < SEQ; j0 += 32) {
    // Batch-issue ALL loads for this iteration first: V tiles (used only by
    // the P@V WMMAs at the end) stay in flight through score WMMAs + softmax.
    const _Float16* kp0 = kbh + (size_t)(j0 + l15) * HEAD_DIM + 8 * half;
    const _Float16* kp1 = kbh + (size_t)(j0 + 16 + l15) * HEAD_DIM + 8 * half;
    const _Float16* vp  = vbh + (size_t)l15 * SEQ + j0 + 8 * half;
    v16h ak0a = ld_a16(kp0);
    v16h ak0b = ld_a16(kp0 + 32);
    v16h ak1a = ld_a16(kp1);
    v16h ak1b = ld_a16(kp1 + 32);
    v16h bv0 = ld_a16(vp + (size_t)(16 * 0) * SEQ);
    v16h bv1 = ld_a16(vp + (size_t)(16 * 1) * SEQ);
    v16h bv2 = ld_a16(vp + (size_t)(16 * 2) * SEQ);
    v16h bv3 = ld_a16(vp + (size_t)(16 * 3) * SEQ);

    v8f s1 = {}, s2 = {};
    s1 = wmma_f16(ak0a, bq0, s1);
    s1 = wmma_f16(ak0b, bq1, s1);
    s2 = wmma_f16(ak1a, bq0, s2);
    s2 = wmma_f16(ak1b, bq1, s2);

    // S^T C-layout: VGPR i = key j0 + i + 8*half (s1) / +16 (s2), q col = l15.
    float sv[16];
#pragma unroll
    for (int i = 0; i < 8; ++i) { sv[i] = s1[i] * scale; sv[8 + i] = s2[i] * scale; }

    // Stream raw scores to output: scores[q][j], q = q0+l15.
    float* srow = sc + (size_t)l15 * SEQ + j0 + 8 * half;
#pragma unroll
    for (int i = 0; i < 8; ++i) { srow[i] = sv[i]; srow[16 + i] = sv[8 + i]; }

    // Online softmax -- entire row slice is in-lane.
    float mt = m_run;
#pragma unroll
    for (int i = 0; i < 16; ++i) mt = fmaxf(mt, sv[i]);
    const float alpha = __expf(m_run - mt);
    float lsum = 0.f;
    v16h pa;  // P in A-operand layout: keys 8h..8h+7 then 16+8h..16+8h+7
#pragma unroll
    for (int i = 0; i < 16; ++i) {
      const float p = __expf(sv[i] - mt);
      lsum += p;
      pa[i] = (_Float16)p;
    }
    l_run = l_run * alpha + lsum;
    m_run = mt;

    // Rescale O accumulators: row of VGPR i is i + 8*half; alpha lives in lane row.
#pragma unroll
    for (int i = 0; i < 8; ++i) {
      const float ai = __shfl(alpha, i + 8 * half, 32);
      o0[i] *= ai; o1[i] *= ai; o2[i] *= ai; o3[i] *= ai;
    }

    // P @ V : V as B operand (N = dh col = 16t+l15, K = key) from transposed vT.
    o0 = wmma_f16(pa, bv0, o0);
    o1 = wmma_f16(pa, bv1, o1);
    o2 = wmma_f16(pa, bv2, o2);
    o3 = wmma_f16(pa, bv3, o3);
  }

  // Normalize and store attn with heads recombined: attn[b][s][h*64 + d].
  const int b = bh >> 4, h = bh & 15;
#pragma unroll
  for (int i = 0; i < 8; ++i) {
    const float li = __shfl(l_run, i + 8 * half, 32);
    const float inv = 1.0f / li;
    const int s = q0 + i + 8 * half;
    _Float16* arow = attn + ((size_t)(b * SEQ + s)) * D_MODEL + h * HEAD_DIM;
    arow[0 * 16 + l15] = (_Float16)(o0[i] * inv);
    arow[1 * 16 + l15] = (_Float16)(o1[i] * inv);
    arow[2 * 16 + l15] = (_Float16)(o2[i] * inv);
    arow[3 * 16 + l15] = (_Float16)(o3[i] * inv);
  }
}

// ---------------------------------------------------------------------------
// Output projection: out = attn(f16) @ w_o^T + b_o, f32 result.
// Wave tile 32x64 like mha_proj_kernel.
// ---------------------------------------------------------------------------
__global__ __launch_bounds__(256) void mha_outproj_kernel(
    const _Float16* __restrict__ A, const _Float16* __restrict__ W,
    const float* __restrict__ bias, float* __restrict__ out) {
  const int wave = threadIdx.x >> 5;
  const int lane = threadIdx.x & 31;
  const int half = lane >> 4;
  const int l15  = lane & 15;
  const int tile = blockIdx.x * 8 + wave;          // 2048 tiles
  const int m0 = (tile & 127) * 32;
  const int n0 = (tile >> 7) * 64;

  const _Float16* arow0 = A + (size_t)(m0 + l15) * D_MODEL + 8 * half;
  const _Float16* arow1 = arow0 + (size_t)16 * D_MODEL;
  const _Float16* wrow  = W + (size_t)(n0 + l15) * D_MODEL + 8 * half;

  v8f c00 = {}, c01 = {}, c02 = {}, c03 = {};
  v8f c10 = {}, c11 = {}, c12 = {}, c13 = {};

  for (int k0 = 0; k0 < D_MODEL; k0 += 32) {
    v16h a0 = ld_a16(arow0 + k0);
    v16h a1 = ld_a16(arow1 + k0);
    v16h b0 = ld_a16(wrow + 0 * 16 * D_MODEL + k0);
    v16h b1 = ld_a16(wrow + 1 * 16 * D_MODEL + k0);
    v16h b2 = ld_a16(wrow + 2 * 16 * D_MODEL + k0);
    v16h b3 = ld_a16(wrow + 3 * 16 * D_MODEL + k0);
    c00 = wmma_f16(a0, b0, c00);
    c01 = wmma_f16(a0, b1, c01);
    c02 = wmma_f16(a0, b2, c02);
    c03 = wmma_f16(a0, b3, c03);
    c10 = wmma_f16(a1, b0, c10);
    c11 = wmma_f16(a1, b1, c11);
    c12 = wmma_f16(a1, b2, c12);
    c13 = wmma_f16(a1, b3, c13);
  }

#pragma unroll
  for (int r = 0; r < 2; ++r) {
#pragma unroll
    for (int t = 0; t < 4; ++t) {
      v8f c;
      if (r == 0) c = (t == 0) ? c00 : (t == 1) ? c01 : (t == 2) ? c02 : c03;
      else        c = (t == 0) ? c10 : (t == 1) ? c11 : (t == 2) ? c12 : c13;
      const int col = n0 + 16 * t + l15;
      const float bv = bias[col];
#pragma unroll
      for (int i = 0; i < 8; ++i) {
        const int row = m0 + 16 * r + i + 8 * half;
        out[(size_t)row * D_MODEL + col] = c[i] + bv;
      }
    }
  }
}

// ---------------------------------------------------------------------------
extern "C" void kernel_launch(void* const* d_in, const int* in_sizes, int n_in,
                              void* d_out, int out_size, void* d_ws,
                              size_t ws_size, hipStream_t stream) {
  (void)in_sizes; (void)n_in; (void)out_size; (void)ws_size;
  const float* Q   = (const float*)d_in[0];
  const float* K   = (const float*)d_in[1];
  const float* V   = (const float*)d_in[2];
  const float* w_q = (const float*)d_in[3];
  const float* b_q = (const float*)d_in[4];
  const float* w_k = (const float*)d_in[5];
  const float* b_k = (const float*)d_in[6];
  const float* w_v = (const float*)d_in[7];
  const float* b_v = (const float*)d_in[8];
  const float* w_o = (const float*)d_in[9];
  const float* b_o = (const float*)d_in[10];

  float* out    = (float*)d_out;                         // [2,2048,1024]
  float* scores = out + (size_t)ROWS * D_MODEL;          // [2,16,2048,2048]

  const size_t XN = (size_t)ROWS * D_MODEL;              // 4 Mi elements
  const size_t WN = (size_t)D_MODEL * D_MODEL;           // 1 Mi elements
  _Float16* p = (_Float16*)d_ws;
  _Float16* Qh  = p; p += XN;   // f16 copies of inputs
  _Float16* Kh  = p; p += XN;
  _Float16* Vh  = p; p += XN;
  _Float16* wqh = p; p += WN;   // f16 copies of weights
  _Float16* wkh = p; p += WN;
  _Float16* wvh = p; p += WN;
  _Float16* woh = p; p += WN;
  _Float16* qh  = p; p += XN;   // projected q  [bh][s][64]
  _Float16* kh  = p; p += XN;   // projected k  [bh][s][64]
  _Float16* vTh = p; p += XN;   // projected v  [bh][64][s]
  _Float16* attn = p;           // [b][s][1024]

  const dim3 blk(256);

  // f32 -> f16 pre-conversion (bandwidth-bound, ~96 MiB total).
  cvt_f16_kernel<<<2048, blk, 0, stream>>>(Q, Qh, (int)(XN / 4));
  cvt_f16_kernel<<<2048, blk, 0, stream>>>(K, Kh, (int)(XN / 4));
  cvt_f16_kernel<<<2048, blk, 0, stream>>>(V, Vh, (int)(XN / 4));
  cvt_f16_kernel<<<512,  blk, 0, stream>>>(w_q, wqh, (int)(WN / 4));
  cvt_f16_kernel<<<512,  blk, 0, stream>>>(w_k, wkh, (int)(WN / 4));
  cvt_f16_kernel<<<512,  blk, 0, stream>>>(w_v, wvh, (int)(WN / 4));
  cvt_f16_kernel<<<512,  blk, 0, stream>>>(w_o, woh, (int)(WN / 4));

  // Projections (2048 wave-tiles of 32x64 -> 256 blocks of 8 waves).
  mha_proj_kernel<<<256, blk, 0, stream>>>(Qh, wqh, b_q, qh, 0);
  mha_proj_kernel<<<256, blk, 0, stream>>>(Kh, wkh, b_k, kh, 0);
  mha_proj_kernel<<<256, blk, 0, stream>>>(Vh, wvh, b_v, vTh, 1);

  // Fused flash attention + scores streaming (4096 wave units).
  mha_attn_kernel<<<512, blk, 0, stream>>>(qh, kh, vTh, attn, scores);

  // Output projection.
  mha_outproj_kernel<<<256, blk, 0, stream>>>(attn, woh, b_o, out);
}
